// CausalSelfAttention_4088808866408
// MI455X (gfx1250) — matrix operand order repaired and checked by reading, not compiled, not run on hardware
//
#include <hip/hip_runtime.h>
#include <hip/hip_bf16.h>

#define BB 4
#define TT 2048
#define CC 1024
#define HH 16
#define DH 64
#define BT (BB*TT)      /* 8192 tokens */
#define C3 (3*CC)       /* 3072 */
#define SCALE 0.125f    /* 1/sqrt(64) */

typedef _Float16 v16h __attribute__((ext_vector_type(16)));
typedef _Float16 v8h  __attribute__((ext_vector_type(8)));
typedef _Float16 v2h  __attribute__((ext_vector_type(2)));
typedef float    v8f  __attribute__((ext_vector_type(8)));

static __device__ __forceinline__ v8f wmma_f16(v16h a, v16h b, v8f c) {
  // D = A(16x32 f16) * B(32x16 f16) + C(16x16 f32)
  return __builtin_amdgcn_wmma_f32_16x16x32_f16(false, a, false, b, (short)0, c, false, false);
}
static __device__ __forceinline__ v16h cat16(v8h lo, v8h hi) {
  return __builtin_shufflevector(lo, hi, 0,1,2,3,4,5,6,7,8,9,10,11,12,13,14,15);
}

// ---------------- fp32 -> f16 convert ----------------
__global__ __launch_bounds__(256) void k_cvt(const float* __restrict__ src,
                                             _Float16* __restrict__ dst, int n) {
  int i = blockIdx.x * 256 + threadIdx.x;
  if (i < n) dst[i] = (_Float16)src[i];
}

// dst[N][K] = (f16) src[K][N]   (transpose + convert)
__global__ __launch_bounds__(256) void k_cvt_t(const float* __restrict__ src,
                                               _Float16* __restrict__ dst, int N, int K) {
  int i = blockIdx.x * 256 + threadIdx.x;
  if (i < N * K) {
    int nn = i / K, kk = i - nn * K;
    dst[i] = (_Float16)src[(size_t)kk * N + nn];
  }
}

// ---------------- QKV GEMM: [8192,1024] x [1024,3072] ----------------
// Each wave computes a 16x64 output tile (4 accumulators), K-loop of 32.
// wt is W_qkv pre-transposed: [3072][1024] f16 -> B-fragments are contiguous.
// Epilogue scatters to Qh[B,H,T,Dh]*SCALE, Kh[B,H,T,Dh], Vt[B,H,Dh,T].
__global__ __launch_bounds__(256) void k_gemm_qkv(const _Float16* __restrict__ xh,
                                                  const _Float16* __restrict__ wt,
                                                  const float* __restrict__ bqkv,
                                                  _Float16* __restrict__ Qh,
                                                  _Float16* __restrict__ Kh,
                                                  _Float16* __restrict__ Vt) {
  int wid  = blockIdx.x * 8 + (threadIdx.x >> 5);
  int lane = threadIdx.x & 31;
  int tn = wid % (C3 / 64);        // 48 column tiles of 64
  int tm = wid / (C3 / 64);        // 512 row tiles of 16
  int r  = lane & 15;
  int hi = lane >> 4;

  const _Float16* arow = xh + (size_t)(tm * 16 + r) * CC;
  v8f acc[4] = {v8f{}, v8f{}, v8f{}, v8f{}};

  for (int kb = 0; kb < CC; kb += 32) {
    if (kb + 32 < CC) __builtin_prefetch(arow + kb + 32, 0, 1);
    v16h a = *(const v16h*)(arow + kb + 16 * hi);
#pragma unroll
    for (int nt = 0; nt < 4; ++nt) {
      v16h b = *(const v16h*)(wt + (size_t)(tn * 64 + nt * 16 + r) * CC + kb + 16 * hi);
      acc[nt] = wmma_f16(a, b, acc[nt]);
    }
  }

#pragma unroll
  for (int nt = 0; nt < 4; ++nt) {
    int col = tn * 64 + nt * 16 + r;           // 0..3071
    float bias = bqkv[col];
    int sec = col >> 10;                        // 0=q 1=k 2=v
    int cc  = col & (CC - 1);
    int h   = cc >> 6;
    int d   = cc & 63;
#pragma unroll
    for (int i = 0; i < 8; ++i) {
      int R  = tm * 16 + i + 8 * hi;            // token index
      int bI = R >> 11;                         // batch
      int t  = R & (TT - 1);
      float val = acc[nt][i] + bias;
      int bh = (bI << 4) + h;
      if (sec == 0)      Qh[((size_t)bh * TT + t) * DH + d] = (_Float16)(val * SCALE);
      else if (sec == 1) Kh[((size_t)bh * TT + t) * DH + d] = (_Float16)val;
      else               Vt[((size_t)bh * DH + d) * TT + t] = (_Float16)val;
    }
  }
}

// ---------------- Flash attention ----------------
// One wave per 16-query tile. Key blocks of 32. Computes S^T = K*Q^T (so each
// lane owns one query column -> softmax reductions are in-lane + one shfl_xor),
// online softmax in fp32, P bounced through wave-private LDS to form the
// A-fragment for O += P*V (V stored transposed -> contiguous B-fragments).
__global__ __launch_bounds__(128) void k_attn(const _Float16* __restrict__ Qh,
                                              const _Float16* __restrict__ Kh,
                                              const _Float16* __restrict__ Vt,
                                              _Float16* __restrict__ yh) {
  __shared__ __align__(32) _Float16 pbuf[4][16][32];
  int wave = threadIdx.x >> 5;
  int lane = threadIdx.x & 31;
  int qt   = blockIdx.x * 4 + wave;
  int bh   = qt >> 7;                 // 128 q-tiles per (b,h)
  int qbase = (qt & 127) << 4;
  int n  = lane & 15;
  int hi = lane >> 4;

  const _Float16* Qb  = Qh + ((size_t)bh * TT + qbase) * DH;
  const _Float16* Kb0 = Kh + (size_t)bh * TT * DH;
  const _Float16* Vb  = Vt + (size_t)bh * DH * TT;
  _Float16* pb = &pbuf[wave][0][0];

  v8f o[4] = {v8f{}, v8f{}, v8f{}, v8f{}};
  float m = -3.0e38f, l = 0.0f;
  int nkb = (qbase >> 5) + 1;         // key blocks needed (causal)

  for (int kb = 0; kb < nkb; ++kb) {
    const _Float16* Kb = Kb0 + (size_t)(kb * 32) * DH;
    v8f s0 = {}, s1 = {};
#pragma unroll
    for (int kk = 0; kk < DH; kk += 32) {
      v8h bql = *(const v8h*)(Qb + (size_t)n * DH + kk + 8 * hi);
      v8h bqh = *(const v8h*)(Qb + (size_t)n * DH + kk + 16 + 8 * hi);
      v16h bq = cat16(bql, bqh);                                        // B = Q^T frag
      v8h a0l = *(const v8h*)(Kb + (size_t)n * DH + kk + 8 * hi);
      v8h a0h = *(const v8h*)(Kb + (size_t)n * DH + kk + 16 + 8 * hi);
      s0 = wmma_f16(cat16(a0l, a0h), bq, s0);                           // keys kb*32+0..15
      v8h a1l = *(const v8h*)(Kb + (size_t)(16 + n) * DH + kk + 8 * hi);
      v8h a1h = *(const v8h*)(Kb + (size_t)(16 + n) * DH + kk + 16 + 8 * hi);
      s1 = wmma_f16(cat16(a1l, a1h), bq, s1);                           // keys kb*32+16..31
    }
    // causal mask: only the last key block can cross the diagonal
    if (kb == nkb - 1) {
#pragma unroll
      for (int i = 0; i < 8; ++i) {
        int k0 = kb * 32 + i + 8 * hi;
        if (k0 > qbase + n)      s0[i] = -3.0e38f;
        if (k0 + 16 > qbase + n) s1[i] = -3.0e38f;
      }
    }
    // online softmax (per query column n; partner lane n^16 holds other keys)
    float mt = -3.0e38f;
#pragma unroll
    for (int i = 0; i < 8; ++i) mt = fmaxf(mt, fmaxf(s0[i], s1[i]));
    mt = fmaxf(mt, __shfl_xor(mt, 16));
    float mnew  = fmaxf(m, mt);
    float alpha = __expf(m - mnew);
    m = mnew;
    float rs = 0.0f;
#pragma unroll
    for (int i = 0; i < 8; ++i) {
      float p0 = __expf(s0[i] - mnew);
      float p1 = __expf(s1[i] - mnew);
      s0[i] = p0; s1[i] = p1; rs += p0 + p1;
    }
    rs += __shfl_xor(rs, 16);
    l = l * alpha + rs;
    // stash P[16 query][32 key] in wave-private LDS (f16)
#pragma unroll
    for (int i = 0; i < 8; i += 2) {
      v2h t0 = { (_Float16)s0[i], (_Float16)s0[i + 1] };
      *(v2h*)(pb + n * 32 + 8 * hi + i) = t0;
      v2h t1 = { (_Float16)s1[i], (_Float16)s1[i + 1] };
      *(v2h*)(pb + n * 32 + 16 + 8 * hi + i) = t1;
    }
    // rescale O rows by alpha(row)
    float arow[8];
#pragma unroll
    for (int i = 0; i < 8; ++i) arow[i] = __shfl(alpha, i + 8 * hi);
#pragma unroll
    for (int nt = 0; nt < 4; ++nt)
#pragma unroll
      for (int i = 0; i < 8; ++i) o[nt][i] *= arow[i];
    // O += P(16x32) * V(32x64): reload P as A-fragment (DS is in-order per wave)
    v16h pa = *(const v16h*)(pb + n * 32 + 16 * hi);
#pragma unroll
    for (int nt = 0; nt < 4; ++nt) {
      v16h bv = *(const v16h*)(Vb + (size_t)(nt * 16 + n) * TT + kb * 32 + 16 * hi);
      o[nt] = wmma_f16(pa, bv, o[nt]);
    }
  }
  // normalize and store to y (f16, [B,T,C] layout = proj A-matrix layout)
  float linv[8];
#pragma unroll
  for (int i = 0; i < 8; ++i) linv[i] = 1.0f / __shfl(l, i + 8 * hi);
  int b = bh >> 4, h = bh & 15;
#pragma unroll
  for (int nt = 0; nt < 4; ++nt)
#pragma unroll
    for (int i = 0; i < 8; ++i) {
      int R   = b * TT + qbase + i + 8 * hi;
      int col = h * DH + nt * 16 + n;
      yh[(size_t)R * CC + col] = (_Float16)(o[nt][i] * linv[i]);
    }
}

// ---------------- Proj GEMM: [8192,1024] x [1024,1024] + bias -> fp32 out ----
__global__ __launch_bounds__(256) void k_gemm_proj(const _Float16* __restrict__ yh,
                                                   const _Float16* __restrict__ wpt,
                                                   const float* __restrict__ bp,
                                                   float* __restrict__ out) {
  int wid  = blockIdx.x * 8 + (threadIdx.x >> 5);
  int lane = threadIdx.x & 31;
  int tn = wid & 15;                  // 16 column tiles of 64
  int tm = wid >> 4;                  // 512 row tiles of 16
  int r  = lane & 15;
  int hi = lane >> 4;

  const _Float16* arow = yh + (size_t)(tm * 16 + r) * CC;
  v8f acc[4] = {v8f{}, v8f{}, v8f{}, v8f{}};

  for (int kb = 0; kb < CC; kb += 32) {
    if (kb + 32 < CC) __builtin_prefetch(arow + kb + 32, 0, 1);
    v16h a = *(const v16h*)(arow + kb + 16 * hi);
#pragma unroll
    for (int nt = 0; nt < 4; ++nt) {
      v16h b = *(const v16h*)(wpt + (size_t)(tn * 64 + nt * 16 + r) * CC + kb + 16 * hi);
      acc[nt] = wmma_f16(a, b, acc[nt]);
    }
  }
#pragma unroll
  for (int nt = 0; nt < 4; ++nt) {
    int col = tn * 64 + nt * 16 + r;
    float bias = bp[col];
#pragma unroll
    for (int i = 0; i < 8; ++i) {
      int R = tm * 16 + i + 8 * hi;
      out[(size_t)R * CC + col] = acc[nt][i] + bias;
    }
  }
}

extern "C" void kernel_launch(void* const* d_in, const int* in_sizes, int n_in,
                              void* d_out, int out_size, void* d_ws, size_t ws_size,
                              hipStream_t stream) {
  const float* x    = (const float*)d_in[0];
  const float* Wqkv = (const float*)d_in[1];
  const float* bqkv = (const float*)d_in[2];
  const float* Wp   = (const float*)d_in[3];
  const float* bp   = (const float*)d_in[4];
  float* out = (float*)d_out;

  char* ws = (char*)d_ws;
  _Float16* xh    = (_Float16*)ws;  ws += (size_t)BT * CC * 2;   // 16.8 MB
  _Float16* wqkvt = (_Float16*)ws;  ws += (size_t)C3 * CC * 2;   //  6.3 MB
  _Float16* wpt   = (_Float16*)ws;  ws += (size_t)CC * CC * 2;   //  2.1 MB
  _Float16* Qh    = (_Float16*)ws;  ws += (size_t)BT * CC * 2;   // 16.8 MB
  _Float16* Kh    = (_Float16*)ws;  ws += (size_t)BT * CC * 2;   // 16.8 MB
  _Float16* Vt    = (_Float16*)ws;  ws += (size_t)BT * CC * 2;   // 16.8 MB
  _Float16* yh    = (_Float16*)ws;  ws += (size_t)BT * CC * 2;   // 16.8 MB

  // 1) convert / pre-transpose to f16
  k_cvt  <<<(BT * CC + 255) / 256, 256, 0, stream>>>(x, xh, BT * CC);
  k_cvt_t<<<(C3 * CC + 255) / 256, 256, 0, stream>>>(Wqkv, wqkvt, C3, CC);
  k_cvt_t<<<(CC * CC + 255) / 256, 256, 0, stream>>>(Wp, wpt, CC, CC);

  // 2) QKV GEMM: 512 m-tiles * 48 n-tiles = 24576 waves / 8 per block
  k_gemm_qkv<<<(512 * 48) / 8, 256, 0, stream>>>(xh, wqkvt, bqkv, Qh, Kh, Vt);

  // 3) flash attention: 64 (b,h) * 128 q-tiles = 8192 waves / 4 per block
  k_attn<<<8192 / 4, 128, 0, stream>>>(Qh, Kh, Vt, yh);

  // 4) output projection: 512 * 16 = 8192 waves / 8 per block
  k_gemm_proj<<<(512 * 16) / 8, 256, 0, stream>>>(yh, wpt, bp, out);
}
